// SPN_90546500534706
// MI455X (gfx1250) — compile-verified, matrix operand order checked by reading, not run on hardware
//
#include <hip/hip_runtime.h>
#include <math.h>

// ---------------------------------------------------------------------------
// SPN forward for MI455X (gfx1250, wave32, WMMA + TDM).
// Convs: implicit GEMM. Block = 256 threads = 8 waves, tile 128px x 64ch.
// Each wave: 32x32 subtile = 2x2 v_wmma_f32_16x16x32_bf16 per 32-wide k-step
// (k-loop unrolled x2 -> 8 static WMMAs). Conv kernel templated on filter
// geometry so im2col address math is mul-shift, not runtime division.
// A (im2col activations, bf16) gathered cooperatively into padded LDS rows;
// B (weights, bf16) moved by the Tensor Data Mover into LDS (padded rows),
// fenced with s_wait_tensorcnt. Propagation: per-line register recurrence.
// ---------------------------------------------------------------------------

typedef __attribute__((ext_vector_type(16))) __bf16   v16bf;
typedef __attribute__((ext_vector_type(16))) short    v16s;
typedef __attribute__((ext_vector_type(8)))  unsigned v8u;
typedef __attribute__((ext_vector_type(8)))  float    v8f;
typedef __attribute__((ext_vector_type(4)))  unsigned v4u;
typedef __attribute__((ext_vector_type(8)))  int      v8i;
typedef __attribute__((ext_vector_type(4)))  int      v4i;

#if defined(__has_builtin)
#if __has_builtin(__builtin_amdgcn_tensor_load_to_lds)
#define HAVE_TDM 1
#endif
#endif
#ifndef HAVE_TDM
#define HAVE_TDM 0
#warning "CDNA5 probe: __builtin_amdgcn_tensor_load_to_lds NOT available - B tile uses cooperative fallback"
#endif

union ABu { v16s s; v8u u; v16bf b; };

__device__ __forceinline__ unsigned short f2bf(float f) {
    union { float f; unsigned u; } x; x.f = f;
    unsigned u = x.u;
    u += 0x7FFFu + ((u >> 16) & 1u);   // RNE to bf16
    return (unsigned short)(u >> 16);
}

// ---------------------------------------------------------------------------
// Implicit-GEMM conv, bf16 in / bf16 weights / f32 accumulate.
// Template on filter geometry -> compile-time divides in the im2col gather.
// act: 0 = none, 2 = ELU (skip added AFTER activation).
// outBf16: 1 -> write bf16 (next conv's input), 0 -> write f32.
// ---------------------------------------------------------------------------
#define BM 128
#define BN 64
#define BK 32
#define ASTR 36   // shorts per A row (72 B: 18 dwords -> conflict-free frags)
#define BSTR 34   // shorts per B row (68 B: 17 dwords, matches TDM pad 16+1)

template <int KH, int KW, int PAD>
__global__ __launch_bounds__(256)
void conv_wmma_bf16(const unsigned short* __restrict__ in,
                    const unsigned short* __restrict__ wgt,
                    const float* __restrict__ bias, const float* __restrict__ skip,
                    void* __restrict__ outp, int outBf16,
                    int NB, int Cin, int Cout, int H, int W, int act)
{
    __shared__ __align__(16) unsigned short a_lds[BM * ASTR];
    __shared__ __align__(16) unsigned short b_lds[BN * BSTR];

    constexpr int KHW = KH * KW;

    const int tid   = threadIdx.x;
    const int lane  = tid & 31;
    const int wid   = tid >> 5;
    const int lh    = lane >> 4;       // K-half of fragment held by this lane
    const int idx16 = lane & 15;
    const int mw    = wid & 3;         // wave's 32-pixel group (0..3)
    const int nw    = wid >> 2;        // wave's 32-channel group (0..1)

    const int HW   = H * W;
    const int M    = NB * HW;
    const int Ktot = Cin * KHW;

    const int m0  = blockIdx.x * BM;
    const int co0 = blockIdx.y * BN;

    __builtin_prefetch(wgt + (size_t)co0 * Ktot, 0, 1);  // global_prefetch_b8

    // this thread's im2col gather row (decoded once)
    const int  mloc = tid & 127;
    const int  kgrp = tid >> 7;        // 0 or 1
    const int  gm   = m0 + mloc;
    const bool gmv  = (gm < M);
    int gnb = 0, gy = 0, gx = 0;
    if (gmv) {
        gnb = gm / HW;
        int r = gm - gnb * HW;
        gy = r / W;
        gx = r - gy * W;
    }
    const unsigned short* inNB = in + (size_t)gnb * Cin * HW;

    v8f acc[2][2] = {};

#pragma unroll 2
    for (int k0 = 0; k0 < Ktot; k0 += BK) {
        __syncthreads();   // protect LDS reuse from previous iteration's reads

#if HAVE_TDM
        if (tid < 32) {
            // B tile via Tensor Data Mover: (Cout x Ktot) bf16 tensor,
            // 64(row) x 32(k) tile at (co0, k0); LDS rows padded to 68 B
            // (pad_interval=3 -> every 16 dwords insert 1 dword).
            unsigned long long ga =
                (unsigned long long)(uintptr_t)(wgt + (size_t)co0 * Ktot + k0);
            unsigned           ldsoff = (unsigned)(uintptr_t)(void*)b_lds;
            unsigned           td0 = (unsigned)(Ktot - k0);   // OOB k -> 0
            unsigned           td1 = (unsigned)(Cout - co0);  // OOB cout -> 0
            unsigned long long st0 = (unsigned long long)(unsigned)Ktot;
            v4u g0;
            g0.x = 1u;                                           // count=1
            g0.y = ldsoff;                                       // lds_addr
            g0.z = (unsigned)ga;                                 // gaddr[31:0]
            g0.w = (unsigned)((ga >> 32) & 0x1FFFFFFu) | (2u << 30); // type=2
            v8i g1;
            g1[0] = (int)((1u << 16) | (1u << 20) | (3u << 22)); // 2B, pad en, ivl=16dw
            g1[1] = (int)((td0 & 0xFFFFu) << 16);
            g1[2] = (int)((td0 >> 16) | ((td1 & 0xFFFFu) << 16));
            g1[3] = (int)((td1 >> 16) | (32u << 16));            // tile_dim0=32
            g1[4] = 64;                                          // tile_dim1=64
            g1[5] = (int)(unsigned)st0;                          // dim0 stride
            g1[6] = (int)((st0 >> 32) & 0xFFFFu);
            g1[7] = 0;
            v4i z4 = {};
#if __clang_major__ >= 23
            v8i z8 = {};
            __builtin_amdgcn_tensor_load_to_lds(g0, g1, z4, z4, z8, 0);
#else
            __builtin_amdgcn_tensor_load_to_lds(g0, g1, z4, z4, 0);
#endif
        }
#else
        // cooperative fallback for the B tile
#pragma unroll
        for (int i = 0; i < (BN * BK) / 256; ++i) {
            int idx = tid + i * 256;
            int k = idx & 31, col = idx >> 5;
            unsigned short v = 0;
            if (co0 + col < Cout && k0 + k < Ktot)
                v = wgt[(size_t)(co0 + col) * Ktot + (k0 + k)];
            b_lds[col * BSTR + k] = v;
        }
#endif

        // A tile: cooperative im2col gather (coalesced along pixels).
        // KHW/KW are compile-time -> divides lower to mul-shift.
#pragma unroll
        for (int i = 0; i < 16; ++i) {
            int k  = kgrp + 2 * i;         // 0..31
            int kk = k0 + k;
            unsigned short v = 0;
            if (gmv && kk < Ktot) {
                int ci = kk / KHW;
                int rr = kk - ci * KHW;
                int ky = rr / KW;
                int kx = rr - ky * KW;
                int iy = gy + ky - PAD;
                int ix = gx + kx - PAD;
                if (iy >= 0 && iy < H && ix >= 0 && ix < W)
                    v = inNB[((size_t)ci * H + iy) * W + ix];
            }
            a_lds[mloc * ASTR + k] = v;
        }

#if HAVE_TDM
        if (tid < 32) __builtin_amdgcn_s_wait_tensorcnt(0);
#endif
        __syncthreads();

        // fragment assembly (ISA 7.12.2 16-bit 16x32 striping) + 2x2 WMMA
        const char* ab = (const char*)a_lds;
        const char* bb = (const char*)b_lds;
        ABu Af[2], Bf[2];
#pragma unroll
        for (int s = 0; s < 2; ++s) {
            int rowA = mw * 32 + s * 16 + idx16;
            int colB = nw * 32 + s * 16 + idx16;
#pragma unroll
            for (int v = 0; v < 8; ++v) {
                int kb = (v < 4) ? (v * 2 + lh * 8) : (16 + (v - 4) * 2 + lh * 8);
                Af[s].u[v] = *(const unsigned*)(ab + rowA * (ASTR * 2) + kb * 2);
                Bf[s].u[v] = *(const unsigned*)(bb + colB * (BSTR * 2) + kb * 2);
            }
        }
#pragma unroll
        for (int s = 0; s < 2; ++s)
#pragma unroll
            for (int s2 = 0; s2 < 2; ++s2)
                acc[s][s2] = __builtin_amdgcn_wmma_f32_16x16x32_bf16(
                    false, Af[s].b, false, Bf[s2].b, (short)0, acc[s][s2],
                    false, false);
    }

    // epilogue: C/D layout -> lane holds N=idx16, rows M = r + 8*lh
    float*          outF = (float*)outp;
    unsigned short* outH = (unsigned short*)outp;
#pragma unroll
    for (int s2 = 0; s2 < 2; ++s2) {
        int co = co0 + nw * 32 + s2 * 16 + idx16;
        if (co >= Cout) continue;
        float bval = bias[co];
#pragma unroll
        for (int s = 0; s < 2; ++s) {
#pragma unroll
            for (int r = 0; r < 8; ++r) {
                int mm = m0 + mw * 32 + s * 16 + r + lh * 8;
                if (mm >= M) continue;
                int nb2 = mm / HW;
                int rem = mm - nb2 * HW;
                float v = acc[s][s2][r] + bval;
                if (act == 2) v = (v > 0.f) ? v : (__expf(v) - 1.f);
                size_t oidx = (size_t)(nb2 * Cout + co) * HW + rem;
                if (skip) v += skip[oidx];
                if (outBf16) outH[oidx] = f2bf(v);
                else         outF[oidx] = v;
            }
        }
    }
}

// ---------------------------------------------------------------------------
// f32 -> bf16 pack
// ---------------------------------------------------------------------------
__global__ void pack_bf16(const float* __restrict__ in,
                          unsigned short* __restrict__ out, int n)
{
    int i = blockIdx.x * blockDim.x + threadIdx.x;
    if (i < n) out[i] = f2bf(in[i]);
}

// ---------------------------------------------------------------------------
// relu + maxpool 3x3 s2 p1 (f32 in, bf16 out: feeds next conv)
// ---------------------------------------------------------------------------
__global__ void relu_maxpool3s2(const float* __restrict__ in,
                                unsigned short* __restrict__ out,
                                int NB, int C, int H, int W)
{
    int Ho = H / 2, Wo = W / 2;
    int total = NB * C * Ho * Wo;
    int i = blockIdx.x * blockDim.x + threadIdx.x;
    if (i >= total) return;
    int ox = i % Wo; int t = i / Wo;
    int oy = t % Ho; t /= Ho;
    int c  = t % C;  int nb = t / C;
    const float* p = in + (size_t)(nb * C + c) * H * W;
    float mx = 0.f;
#pragma unroll
    for (int dy = 0; dy < 3; ++dy) {
        int iy = oy * 2 - 1 + dy;
        if (iy < 0 || iy >= H) continue;
#pragma unroll
        for (int dx = 0; dx < 3; ++dx) {
            int ix = ox * 2 - 1 + dx;
            if (ix < 0 || ix >= W) continue;
            float v = p[(size_t)iy * W + ix];
            v = v > 0.f ? v : 0.f;
            mx = fmaxf(mx, v);
        }
    }
    out[i] = f2bf(mx);
}

// ---------------------------------------------------------------------------
// bilinear x2 upsample, half-pixel centers (f32 in, bf16 out)
// ---------------------------------------------------------------------------
__global__ void up2_bilinear(const float* __restrict__ in,
                             unsigned short* __restrict__ out,
                             int NB, int C, int H, int W)
{
    int Ho = 2 * H, Wo = 2 * W;
    int total = NB * C * Ho * Wo;
    int i = blockIdx.x * blockDim.x + threadIdx.x;
    if (i >= total) return;
    int ox = i % Wo; int t = i / Wo;
    int oy = t % Ho; t /= Ho;
    int c  = t % C;  int nb = t / C;

    float sy = (oy + 0.5f) * 0.5f - 0.5f;
    float sx = (ox + 0.5f) * 0.5f - 0.5f;
    int y0 = (int)floorf(sy); float fy = sy - y0;
    int x0 = (int)floorf(sx); float fx = sx - x0;
    int y0c = min(max(y0, 0), H - 1), y1c = min(max(y0 + 1, 0), H - 1);
    int x0c = min(max(x0, 0), W - 1), x1c = min(max(x0 + 1, 0), W - 1);

    const float* p = in + (size_t)(nb * C + c) * H * W;
    float v = (1.f - fy) * ((1.f - fx) * p[(size_t)y0c * W + x0c] + fx * p[(size_t)y0c * W + x1c])
            +        fy  * ((1.f - fx) * p[(size_t)y1c * W + x0c] + fx * p[(size_t)y1c * W + x1c]);
    out[i] = f2bf(v);
}

// ---------------------------------------------------------------------------
// Directional gated propagation: one wave per 192-element line (6/lane),
// 96 recurrence steps with lane-shuffle carries and edge replication.
// ---------------------------------------------------------------------------
#define PROP_PL 6

__global__ __launch_bounds__(32)
void propagate_dir(const float* __restrict__ X, const float* __restrict__ guide,
                   float* __restrict__ out, int NB, int C, int H, int W,
                   int dir, int n_steps)
{
    const int lane  = threadIdx.x & 31;
    const bool horiz = (dir < 2);
    const bool rev   = (dir & 1);
    const int nfix  = horiz ? H : W;

    int line = blockIdx.x;
    int fix  = line % nfix; int t = line / nfix;
    int c    = t % C;       int nb = t / C;

    const int GC = 12 * C;
    const size_t HW = (size_t)H * W;
    size_t planeX = (size_t)(nb * C + c) * HW;
    size_t g1p = ((size_t)nb * GC + (size_t)(3 * dir + 0) * C + c) * HW;
    size_t g2p = g1p + (size_t)C * HW;
    size_t g3p = g2p + (size_t)C * HW;

    size_t base0; int stride;
    if (horiz) { base0 = (size_t)fix * W; stride = 1; }
    else       { base0 = (size_t)fix;     stride = W; }

    float bse[PROP_PL], g2v[PROP_PL], o[PROP_PL];
#pragma unroll
    for (int j = 0; j < PROP_PL; ++j) {
        int i = lane * PROP_PL + j;
        size_t off = base0 + (size_t)i * stride;
        float g1 = guide[g1p + off];
        float g2 = guide[g2p + off];
        float g3 = guide[g3p + off];
        float s = fabsf(g1) + fabsf(g2) + fabsf(g3);
        s = (s == 0.f) ? 1e-6f : s;
        if (s >= 1.f) { g1 /= s; g2 /= s; }
        bse[j] = g1 * X[planeX + off];
        g2v[j] = g2;
        o[j]   = bse[j];
    }

    for (int it = 0; it < n_steps; ++it) {
        float nw[PROP_PL];
        if (!rev) {
            float carry = __shfl_up(o[PROP_PL - 1], 1);
            float sh = (lane == 0) ? o[0] : carry;
            nw[0] = bse[0] + g2v[0] * sh;
#pragma unroll
            for (int j = 1; j < PROP_PL; ++j) nw[j] = bse[j] + g2v[j] * o[j - 1];
        } else {
            float carry = __shfl_down(o[0], 1);
            float sh = (lane == 31) ? o[PROP_PL - 1] : carry;
            nw[PROP_PL - 1] = bse[PROP_PL - 1] + g2v[PROP_PL - 1] * sh;
#pragma unroll
            for (int j = 0; j < PROP_PL - 1; ++j) nw[j] = bse[j] + g2v[j] * o[j + 1];
        }
#pragma unroll
        for (int j = 0; j < PROP_PL; ++j) o[j] = nw[j];
    }

#pragma unroll
    for (int j = 0; j < PROP_PL; ++j) {
        int i = lane * PROP_PL + j;
        out[planeX + base0 + (size_t)i * stride] = o[j];
    }
}

// ---------------------------------------------------------------------------
// elementwise max of 4 (f32 in, bf16 out: feeds post conv)
// ---------------------------------------------------------------------------
__global__ void max4(const float* __restrict__ a, const float* __restrict__ b,
                     const float* __restrict__ c, const float* __restrict__ d,
                     unsigned short* __restrict__ out, int n)
{
    int i = blockIdx.x * blockDim.x + threadIdx.x;
    if (i >= n) return;
    out[i] = f2bf(fmaxf(fmaxf(a[i], b[i]), fmaxf(c[i], d[i])));
}

// ---------------------------------------------------------------------------
// host-side orchestration
// ---------------------------------------------------------------------------
extern "C" void kernel_launch(void* const* d_in, const int* in_sizes, int n_in,
                              void* d_out, int out_size, void* d_ws, size_t ws_size,
                              hipStream_t stream)
{
    (void)in_sizes; (void)n_in; (void)out_size; (void)ws_size;

    const float* x      = (const float*)d_in[0];
    const float* rgb    = (const float*)d_in[1];
    const float* mask_w = (const float*)d_in[2];
    const float* mask_b = (const float*)d_in[3];
    const float* vgg_w1 = (const float*)d_in[4];
    const float* vgg_b1 = (const float*)d_in[5];
    const float* vgg_w2 = (const float*)d_in[6];
    const float* vgg_b2 = (const float*)d_in[7];
    const float* vgg_w3 = (const float*)d_in[8];
    const float* vgg_b3 = (const float*)d_in[9];
    const float* vgg_w4 = (const float*)d_in[10];
    const float* vgg_b4 = (const float*)d_in[11];
    const float* dec_w0 = (const float*)d_in[12];
    const float* dec_b0 = (const float*)d_in[13];
    const float* dec_w2 = (const float*)d_in[14];
    const float* dec_b2 = (const float*)d_in[15];
    const float* dec_w4 = (const float*)d_in[16];
    const float* dec_b4 = (const float*)d_in[17];
    const float* dec_w6 = (const float*)d_in[18];
    const float* dec_b6 = (const float*)d_in[19];
    const float* dec_w7 = (const float*)d_in[20];
    const float* dec_b7 = (const float*)d_in[21];
    const float* post_w = (const float*)d_in[22];
    const float* post_b = (const float*)d_in[23];

    const int NB = 2, NF = 32, H = 192, W = 192;

    // --- workspace carve-out (byte bump allocator, 256-B aligned) ---
    char* wsb = (char*)d_ws;
    size_t off = 0;
    auto allocB = [&](size_t bytes) -> void* {
        void* p = wsb + off;
        off += (bytes + 255) & ~(size_t)255;
        return p;
    };
    auto allocF = [&](size_t n) { return (float*)allocB(n * 4); };
    auto allocH = [&](size_t n) { return (unsigned short*)allocB(n * 2); };

    // f32 activations (consumed by pools / skips / propagation)
    float* X     = allocF((size_t)NB * NF      * 192 * 192);
    float* c1    = allocF((size_t)NB * NF      * 192 * 192);
    float* c2    = allocF((size_t)NB * NF * 2  *  96 *  96);
    float* c3    = allocF((size_t)NB * NF * 4  *  48 *  48);
    float* d0    = allocF((size_t)NB * NF * 4  *  24 *  24);
    float* d1    = allocF((size_t)NB * NF * 4  *  48 *  48);
    float* d2    = allocF((size_t)NB * NF * 2  *  96 *  96);
    float* guide = allocF((size_t)NB * NF * 12 * 192 * 192);
    float* o1    = allocF((size_t)NB * NF      * 192 * 192);
    float* o2    = allocF((size_t)NB * NF      * 192 * 192);
    float* o3    = allocF((size_t)NB * NF      * 192 * 192);
    float* o4    = allocF((size_t)NB * NF      * 192 * 192);

    // bf16 activations (conv inputs)
    unsigned short* xb    = allocH((size_t)NB * 3      * 192 * 192);
    unsigned short* rgbb  = allocH((size_t)NB * 3      * 192 * 192);
    unsigned short* p1b   = allocH((size_t)NB * NF     *  96 *  96);
    unsigned short* p2b   = allocH((size_t)NB * NF * 2 *  48 *  48);
    unsigned short* p3b   = allocH((size_t)NB * NF * 4 *  24 *  24);
    unsigned short* c4b   = allocH((size_t)NB * NF * 8 *  24 *  24);
    unsigned short* u1b   = allocH((size_t)NB * NF * 4 *  48 *  48);
    unsigned short* u2b   = allocH((size_t)NB * NF * 4 *  96 *  96);
    unsigned short* u3b   = allocH((size_t)NB * NF * 2 * 192 * 192);
    unsigned short* d3b   = allocH((size_t)NB * NF     * 192 * 192);
    unsigned short* omaxb = allocH((size_t)NB * NF     * 192 * 192);

    // bf16 weights
    unsigned short* wb_mask = allocH((size_t)NF * 27);
    unsigned short* wb_v1   = allocH((size_t)NF * 27);
    unsigned short* wb_v2   = allocH((size_t)NF * 2 * NF * 9);
    unsigned short* wb_v3   = allocH((size_t)NF * 4 * NF * 2 * 9);
    unsigned short* wb_v4   = allocH((size_t)NF * 8 * NF * 4 * 9);
    unsigned short* wb_d0   = allocH((size_t)NF * 4 * NF * 8);
    unsigned short* wb_d2   = allocH((size_t)NF * 4 * NF * 4 * 9);
    unsigned short* wb_d4   = allocH((size_t)NF * 2 * NF * 4 * 9);
    unsigned short* wb_d6   = allocH((size_t)NF * NF * 2 * 9);
    unsigned short* wb_d7   = allocH((size_t)NF * 12 * NF * 9);
    unsigned short* wb_post = allocH((size_t)3 * NF * 9);

    auto pack = [&](const float* src, unsigned short* dst, size_t n) {
        pack_bf16<<<((int)n + 255) / 256, 256, 0, stream>>>(src, dst, (int)n);
    };
    auto conv3 = [&](const unsigned short* in, const unsigned short* wg,
                     const float* bi, const float* sk, void* out, int outBf16,
                     int Cin, int Cout, int Hh, int Ww, int act) {
        int M = NB * Hh * Ww;
        dim3 g((M + BM - 1) / BM, (Cout + BN - 1) / BN);
        conv_wmma_bf16<3, 3, 1><<<g, 256, 0, stream>>>(in, wg, bi, sk, out, outBf16,
                                                       NB, Cin, Cout, Hh, Ww, act);
    };
    auto conv1 = [&](const unsigned short* in, const unsigned short* wg,
                     const float* bi, void* out, int outBf16,
                     int Cin, int Cout, int Hh, int Ww) {
        int M = NB * Hh * Ww;
        dim3 g((M + BM - 1) / BM, (Cout + BN - 1) / BN);
        conv_wmma_bf16<1, 1, 0><<<g, 256, 0, stream>>>(in, wg, bi, nullptr, out,
                                                       outBf16, NB, Cin, Cout,
                                                       Hh, Ww, 0);
    };
    auto pool = [&](const float* in, unsigned short* out, int C, int Hh) {
        int total = NB * C * (Hh / 2) * (Hh / 2);
        relu_maxpool3s2<<<(total + 255) / 256, 256, 0, stream>>>(in, out, NB, C, Hh, Hh);
    };
    auto up = [&](const float* in, unsigned short* out, int C, int Hh) {
        int total = NB * C * 4 * Hh * Hh;
        up2_bilinear<<<(total + 255) / 256, 256, 0, stream>>>(in, out, NB, C, Hh, Hh);
    };

    // one-time bf16 packs (inputs + weights)
    pack(x,      xb,      (size_t)NB * 3 * 192 * 192);
    pack(rgb,    rgbb,    (size_t)NB * 3 * 192 * 192);
    pack(mask_w, wb_mask, (size_t)NF * 27);
    pack(vgg_w1, wb_v1,   (size_t)NF * 27);
    pack(vgg_w2, wb_v2,   (size_t)NF * 2 * NF * 9);
    pack(vgg_w3, wb_v3,   (size_t)NF * 4 * NF * 2 * 9);
    pack(vgg_w4, wb_v4,   (size_t)NF * 8 * NF * 4 * 9);
    pack(dec_w0, wb_d0,   (size_t)NF * 4 * NF * 8);
    pack(dec_w2, wb_d2,   (size_t)NF * 4 * NF * 4 * 9);
    pack(dec_w4, wb_d4,   (size_t)NF * 2 * NF * 4 * 9);
    pack(dec_w6, wb_d6,   (size_t)NF * NF * 2 * 9);
    pack(dec_w7, wb_d7,   (size_t)NF * 12 * NF * 9);
    pack(post_w, wb_post, (size_t)3 * NF * 9);

    // X = mask conv
    conv3(xb, wb_mask, mask_b, nullptr, X, 0, 3, NF, 192, 192, 0);

    // VGG encoder
    conv3(rgbb, wb_v1, vgg_b1, nullptr, c1, 0, 3, NF, 192, 192, 0);
    pool(c1, p1b, NF, 192);
    conv3(p1b, wb_v2, vgg_b2, nullptr, c2, 0, NF, NF * 2, 96, 96, 0);
    pool(c2, p2b, NF * 2, 96);
    conv3(p2b, wb_v3, vgg_b3, nullptr, c3, 0, NF * 2, NF * 4, 48, 48, 0);
    pool(c3, p3b, NF * 4, 48);
    conv3(p3b, wb_v4, vgg_b4, nullptr, c4b, 1, NF * 4, NF * 8, 24, 24, 0);

    // Decoder
    conv1(c4b, wb_d0, dec_b0, d0, 0, NF * 8, NF * 4, 24, 24);
    up(d0, u1b, NF * 4, 24);
    conv3(u1b, wb_d2, dec_b2, c3, d1, 0, NF * 4, NF * 4, 48, 48, 2);
    up(d1, u2b, NF * 4, 48);
    conv3(u2b, wb_d4, dec_b4, c2, d2, 0, NF * 4, NF * 2, 96, 96, 2);
    up(d2, u3b, NF * 2, 96);
    conv3(u3b, wb_d6, dec_b6, c1, d3b, 1, NF * 2, NF, 192, 192, 2);
    conv3(d3b, wb_d7, dec_b7, nullptr, guide, 0, NF, NF * 12, 192, 192, 0);

    // Directional propagation (n_steps = max(H,W)/2 = 96)
    const int n_steps = 96;
    const int lines = NB * NF * 192;
    float* outs[4] = { o1, o2, o3, o4 };
    for (int dir = 0; dir < 4; ++dir)
        propagate_dir<<<lines, 32, 0, stream>>>(X, guide, outs[dir], NB, NF, H, W,
                                                dir, n_steps);

    int nelem = NB * NF * H * W;
    max4<<<(nelem + 255) / 256, 256, 0, stream>>>(o1, o2, o3, o4, omaxb, nelem);

    // Post conv -> output (2,3,192,192) f32
    conv3(omaxb, wb_post, post_b, nullptr, d_out, 0, NF, 3, 192, 192, 0);
}
